// SelfAttention_5342939317070
// MI455X (gfx1250) — matrix operand order repaired and checked by reading, not compiled, not run on hardware
//
#include <hip/hip_runtime.h>

typedef _Float16 fp16;
typedef __attribute__((ext_vector_type(16))) _Float16 v16h;
typedef __attribute__((ext_vector_type(8)))  _Float16 v8h;
typedef __attribute__((ext_vector_type(8)))  float    v8f;
typedef __attribute__((ext_vector_type(4)))  float    v4f;

#define B_ 4
#define S_ 4096
#define D_ 256
#define A_ 64
#define O_ 256
#define NW 8
#define ROWS_PER_BLOCK (16*NW)

union V16U { v16h v; v8h h[2]; };

__device__ inline v16h ld16(const fp16* p) {
  V16U u; u.h[0] = *(const v8h*)(p); u.h[1] = *(const v8h*)(p + 8); return u.v;
}
__device__ inline v16h ld16_split(const fp16* p0, const fp16* p1) {
  V16U u; u.h[0] = *(const v8h*)(p0); u.h[1] = *(const v8h*)(p1); return u.v;
}

// ---------------- weight prep: fp32 -> f16, transposed ----------------
__global__ void prep_weights(const float* __restrict__ Wk, const float* __restrict__ Wq,
                             const float* __restrict__ Wv, const float* __restrict__ Wo,
                             fp16* wkT, fp16* wqT, fp16* wvT, fp16* woT) {
  int i = blockIdx.x * 256 + threadIdx.x;      // 0..16383
  {
    int n = i >> 8, k = i & 255;               // wT[n(64)][k(256)] = W[k][n]
    wkT[i] = (fp16)Wk[k * A_ + n];
    wqT[i] = (fp16)Wq[k * A_ + n];
    wvT[i] = (fp16)Wv[k * A_ + n];
  }
  {
    int n = i >> 6, k = i & 63;                // woT[n(256)][k(64)] = Wo[k][n]
    woT[i] = (fp16)Wo[k * O_ + n];
  }
}

// ---------------- projections: k,q,v = x @ W (WMMA f16) ----------------
__global__ void __launch_bounds__(256)
proj_kernel(const float* __restrict__ x, const fp16* __restrict__ wkT,
            const fp16* __restrict__ wqT, const fp16* __restrict__ wvT,
            fp16* kh, fp16* qh, fp16* vTh) {
  const int b    = blockIdx.y;
  const int wave = threadIdx.x >> 5;
  const int lane = threadIdx.x & 31;
  const int g    = lane >> 4, ln = lane & 15;
  const int s0   = blockIdx.x * ROWS_PER_BLOCK + wave * 16;

  const float* xr = x + ((size_t)(b * S_) + s0 + ln) * D_;

  v8f ck[4] = {}, cq[4] = {}, cv[4] = {};
  for (int kc = 0; kc < 8; ++kc) {
    int col0 = kc * 32 + g * 16;
    v4f f0 = *(const v4f*)(xr + col0);
    v4f f1 = *(const v4f*)(xr + col0 + 4);
    v4f f2 = *(const v4f*)(xr + col0 + 8);
    v4f f3 = *(const v4f*)(xr + col0 + 12);
    V16U a;
    for (int t = 0; t < 4; ++t) {
      a.h[0][t]     = (fp16)f0[t];
      a.h[0][t + 4] = (fp16)f1[t];
      a.h[1][t]     = (fp16)f2[t];
      a.h[1][t + 4] = (fp16)f3[t];
    }
    for (int nc = 0; nc < 4; ++nc) {
      int wof = (nc * 16 + ln) * D_ + kc * 32 + g * 16;
      v16h bk = ld16(wkT + wof);
      ck[nc] = __builtin_amdgcn_wmma_f32_16x16x32_f16(false, a.v, false, bk, (short)0, ck[nc], false, false);
      v16h bq = ld16(wqT + wof);
      cq[nc] = __builtin_amdgcn_wmma_f32_16x16x32_f16(false, a.v, false, bq, (short)0, cq[nc], false, false);
      v16h bv = ld16(wvT + wof);
      cv[nc] = __builtin_amdgcn_wmma_f32_16x16x32_f16(false, a.v, false, bv, (short)0, cv[nc], false, false);
    }
  }

  const float scale = 0.125f;  // 1/sqrt(A)
  for (int nc = 0; nc < 4; ++nc) {
    int n = nc * 16 + ln;
    v8h vpack = {};
    for (int r = 0; r < 8; ++r) {
      int srow = s0 + r + 8 * g;
      size_t base = ((size_t)(b * S_) + srow) * A_ + n;
      kh[base] = (fp16)(ck[nc][r] * scale);   // pre-scale K so logits come out scaled
      qh[base] = (fp16)cq[nc][r];
      vpack[r] = (fp16)cv[nc][r];
    }
    // vT stored [b][a][s]: consecutive r -> consecutive s -> one 16B store
    *(v8h*)(vTh + ((size_t)(b * A_) + n) * S_ + s0 + 8 * g) = vpack;
  }
}

// ---------------- flash attention + fused @Wo ----------------
__global__ void __launch_bounds__(256)
attn_kernel(const fp16* __restrict__ kh, const fp16* __restrict__ qh,
            const fp16* __restrict__ vTh, const fp16* __restrict__ woT,
            float* __restrict__ out) {
  __shared__ fp16 q_lds[32 * 64];       // [t_local][a]
  __shared__ fp16 v_lds[64 * 32];       // [a][t_local]
  __shared__ fp16 p_lds[NW][16 * 32];   // per-wave [m][t_local]
  __shared__ fp16 st_lds[NW][16 * 64];  // per-wave attn staging [m][a]

  const int b    = blockIdx.y;
  const int wave = threadIdx.x >> 5;
  const int lane = threadIdx.x & 31;
  const int g    = lane >> 4, ln = lane & 15;
  const int s0   = blockIdx.x * ROWS_PER_BLOCK + wave * 16;
  const int tid  = threadIdx.x;

  // A-fragments of K rows (these are the "query" rows: logits[s,t] = k[s].q[t])
  const fp16* krow = kh + ((size_t)(b * S_) + s0 + ln) * A_;
  v16h ka0 = ld16(krow + g * 16);
  v16h ka1 = ld16(krow + 32 + g * 16);

  float rowm[8], rowl[8];
  v8f acc[4] = {};
  for (int r = 0; r < 8; ++r) { rowm[r] = -1e30f; rowl[r] = 0.f; }

  for (int it = 0; it < S_ / 32; ++it) {
    const int t0 = it * 32;
    __syncthreads();
    {  // q tile [32][64]
      int idx = tid * 8, tl = idx >> 6, a0 = idx & 63;
      *(v8h*)(q_lds + idx) = *(const v8h*)(qh + ((size_t)(b * S_) + t0 + tl) * A_ + a0);
    }
    {  // vT tile [64][32]
      int idx = tid * 8, aa = idx >> 5, tt = idx & 31;
      *(v8h*)(v_lds + idx) = *(const v8h*)(vTh + ((size_t)(b * A_) + aa) * S_ + t0 + tt);
    }
    __syncthreads();

    // logits: 16 (s) x 32 (t), inner dim a=64 -> 4 WMMAs
    v8f sfr[2];
    for (int tc = 0; tc < 2; ++tc) {
      v8f c = {};
      const fp16* qb = q_lds + (tc * 16 + ln) * 64 + g * 16;
      v16h b0 = ld16(qb);
      c = __builtin_amdgcn_wmma_f32_16x16x32_f16(false, ka0, false, b0, (short)0, c, false, false);
      v16h b1 = ld16(qb + 32);
      c = __builtin_amdgcn_wmma_f32_16x16x32_f16(false, ka1, false, b1, (short)0, c, false, false);
      sfr[tc] = c;
    }

    // online softmax (row stats reduced over the 16-lane half)
    for (int r = 0; r < 8; ++r) {
      float v0 = sfr[0][r], v1 = sfr[1][r];
      float mx = fmaxf(v0, v1);
      for (int off = 1; off < 16; off <<= 1) mx = fmaxf(mx, __shfl_xor(mx, off, 16));
      float mnew  = fmaxf(rowm[r], mx);
      float alpha = __expf(rowm[r] - mnew);
      float p0 = __expf(v0 - mnew);
      float p1 = __expf(v1 - mnew);
      float ssum = p0 + p1;
      for (int off = 1; off < 16; off <<= 1) ssum += __shfl_xor(ssum, off, 16);
      rowl[r] = rowl[r] * alpha + ssum;
      rowm[r] = mnew;
      for (int nc = 0; nc < 4; ++nc) acc[nc][r] = acc[nc][r] * alpha;
      int m = r + 8 * g;
      p_lds[wave][m * 32 + ln]      = (fp16)p0;
      p_lds[wave][m * 32 + 16 + ln] = (fp16)p1;
    }
    asm volatile("s_wait_dscnt 0" ::: "memory");

    // P (16x32) as one A fragment; acc += P @ V
    v16h pA = ld16(&p_lds[wave][ln * 32 + g * 16]);
    for (int nc = 0; nc < 4; ++nc) {
      v16h bv = ld16(v_lds + (nc * 16 + ln) * 32 + g * 16);
      acc[nc] = __builtin_amdgcn_wmma_f32_16x16x32_f16(false, pA, false, bv, (short)0, acc[nc], false, false);
    }
  }

  // epilogue: attn = acc / l, then out = attn @ Wo (fused)
  for (int r = 0; r < 8; ++r) {
    float inv = 1.f / rowl[r];
    int m = r + 8 * g;
    for (int nc = 0; nc < 4; ++nc)
      st_lds[wave][m * 64 + nc * 16 + ln] = (fp16)(acc[nc][r] * inv);
  }
  asm volatile("s_wait_dscnt 0" ::: "memory");

  v16h aa0 = ld16(&st_lds[wave][ln * 64 + g * 16]);
  v16h aa1 = ld16(&st_lds[wave][ln * 64 + 32 + g * 16]);

  for (int n16 = 0; n16 < 16; ++n16) {
    v8f c = {};
    const fp16* wb = woT + (n16 * 16 + ln) * A_ + g * 16;
    v16h b0 = ld16(wb);
    c = __builtin_amdgcn_wmma_f32_16x16x32_f16(false, aa0, false, b0, (short)0, c, false, false);
    v16h b1 = ld16(wb + 32);
    c = __builtin_amdgcn_wmma_f32_16x16x32_f16(false, aa1, false, b1, (short)0, c, false, false);
    for (int r = 0; r < 8; ++r)
      out[((size_t)(b * S_) + s0 + r + 8 * g) * O_ + n16 * 16 + ln] = c[r];
  }
}

// ---------------- launch ----------------
extern "C" void kernel_launch(void* const* d_in, const int* in_sizes, int n_in,
                              void* d_out, int out_size, void* d_ws, size_t ws_size,
                              hipStream_t stream) {
  (void)in_sizes; (void)n_in; (void)out_size; (void)ws_size;
  const float* x  = (const float*)d_in[0];
  const float* Wk = (const float*)d_in[1];
  const float* Wq = (const float*)d_in[2];
  const float* Wv = (const float*)d_in[3];
  const float* Wo = (const float*)d_in[4];
  float* out = (float*)d_out;

  fp16* ws  = (fp16*)d_ws;
  fp16* wkT = ws;                                  // 64*256
  fp16* wqT = wkT + 64 * 256;
  fp16* wvT = wqT + 64 * 256;
  fp16* woT = wvT + 64 * 256;                      // 256*64
  fp16* kh  = woT + 256 * 64;                      // B*S*A
  fp16* qh  = kh + (size_t)B_ * S_ * A_;
  fp16* vTh = qh + (size_t)B_ * S_ * A_;           // stored [b][a][s]

  prep_weights<<<64, 256, 0, stream>>>(Wk, Wq, Wv, Wo, wkT, wqT, wvT, woT);
  proj_kernel<<<dim3(S_ / ROWS_PER_BLOCK, B_), 256, 0, stream>>>(x, wkT, wqT, wvT, kh, qh, vTh);
  attn_kernel<<<dim3(S_ / ROWS_PER_BLOCK, B_), 256, 0, stream>>>(kh, qh, vTh, woT, out);
}
